// AMK_Block_52845277610238
// MI455X (gfx1250) — matrix operand order repaired and checked by reading, not compiled, not run on hardware
//
#include <hip/hip_runtime.h>

#define Bn 4
#define Nn 4096
#define DM 1024
#define DS 64
#define INNER 4096
#define G2 8192

typedef __attribute__((ext_vector_type(16))) _Float16 v16h;
typedef __attribute__((ext_vector_type(8)))  float    v8f;

#define WMMA_F16(a, b, c) \
    __builtin_amdgcn_wmma_f32_16x16x32_f16(false, (a), false, (b), (short)0, (c), false, false)

// ---------------------------------------------------------------------------
// WMMA fragment layout helpers (CDNA5 ISA 7.12.2, wave32, 16-bit operands)
// A tile: 16x32 (MxK). lane = m + 16*hi where hi=(k&8)>>3;
//   k' = k-8*hi; v = k'<8 ? k'>>1 : 4+((k'-16)>>1); j = 2v + (k'&1)
// B tile: 32x16 (KxN). lane = n + 16*((k>>3)&1); j = (k&7) + 8*(k>>4)
// C tile: 16x16 f32, reg i, lane l -> m = i + 8*(l>>4), n = l&15
// Tiles stored as 512 contiguous halfs: idx = tile*512 + lane*16 + j
// ---------------------------------------------------------------------------
__device__ __forceinline__ void a_frag_pos(int m, int k, int& lane, int& j) {
    int hi = (k >> 3) & 1;
    int kp = k - (hi << 3);
    int v  = (kp < 8) ? (kp >> 1) : (4 + ((kp - 16) >> 1));
    lane = m + (hi << 4);
    j    = (v << 1) | (kp & 1);
}
__device__ __forceinline__ void b_frag_pos(int k, int n, int& lane, int& j) {
    lane = n + (((k >> 3) & 1) << 4);
    j    = (k & 7) + ((k >> 4) << 3);
}
__device__ __forceinline__ void write_afrag(_Float16* F, int m, int k, int Kt, float v) {
    int fl, fj; a_frag_pos(m & 15, k & 31, fl, fj);
    F[((size_t)((m >> 4) * Kt + (k >> 5)) << 9) + fl * 16 + fj] = (_Float16)v;
}
__device__ __forceinline__ void write_bfrag(_Float16* F, int k, int n, int Nt, float v) {
    int fl, fj; b_frag_pos(k & 31, n & 15, fl, fj);
    F[((size_t)((k >> 5) * Nt + (n >> 4)) << 9) + fl * 16 + fj] = (_Float16)v;
}
__device__ __forceinline__ float read_afrag(const _Float16* F, int m, int k, int Kt) {
    int fl, fj; a_frag_pos(m & 15, k & 31, fl, fj);
    return (float)F[((size_t)((m >> 4) * Kt + (k >> 5)) << 9) + fl * 16 + fj];
}
__device__ __forceinline__ v16h load_frag(const _Float16* F, size_t tile, int lane) {
    return *(const v16h*)(F + (tile << 9) + lane * 16);
}

// ---------------------------------------------------------------------------
// Kernel 1: LayerNorm1 + RoPE. Writes Q_norm1 (f32 row-major), V as B-frags,
// Q_rot as A-frags.
// ---------------------------------------------------------------------------
__global__ void ln_rope_kernel(const float* __restrict__ Qin,
                               const float* __restrict__ cosT,
                               const float* __restrict__ sinT,
                               const float* __restrict__ gw,
                               const float* __restrict__ gb,
                               float* __restrict__ Qn1,
                               _Float16* __restrict__ Vb,
                               _Float16* __restrict__ QrotA) {
    __shared__ float red[256];
    __shared__ float row[DM];
    int n = blockIdx.x, t = threadIdx.x;
    const float* src = Qin + (size_t)n * DM;
    float x[4], s = 0.f;
    for (int i = 0; i < 4; ++i) { x[i] = src[t + i * 256]; s += x[i]; }
    red[t] = s; __syncthreads();
    for (int o = 128; o > 0; o >>= 1) { if (t < o) red[t] += red[t + o]; __syncthreads(); }
    float mu = red[0] * (1.f / DM); __syncthreads();
    float s2 = 0.f;
    for (int i = 0; i < 4; ++i) { float d = x[i] - mu; s2 += d * d; }
    red[t] = s2; __syncthreads();
    for (int o = 128; o > 0; o >>= 1) { if (t < o) red[t] += red[t + o]; __syncthreads(); }
    float inv = rsqrtf(red[0] * (1.f / DM) + 1e-5f); __syncthreads();
    for (int i = 0; i < 4; ++i) {
        int d = t + i * 256;
        float y = (x[i] - mu) * inv * gw[d] + gb[d];
        row[d] = y;
        Qn1[(size_t)n * DM + d] = y;
        write_bfrag(Vb, n, d, DM / 16, y);          // V B-frag: K=n, N=d
    }
    __syncthreads();
    for (int i = 0; i < 4; ++i) {
        int d = t + i * 256;
        float part = (d < DM / 2) ? -row[d + DM / 2] : row[d - DM / 2];
        float r = row[d] * cosT[(size_t)n * DM + d] + part * sinT[(size_t)n * DM + d];
        write_afrag(QrotA, n, d, DM / 32, r);        // Qrot A-frag: M=n, K=d
    }
}

// Kernel 2: q_pool = mean over N of Q_norm1
__global__ void pool_kernel(const float* __restrict__ Qn1, float* __restrict__ qpool) {
    int d = blockIdx.x * 256 + threadIdx.x;
    float s = 0.f;
    for (int n = 0; n < Nn; ++n) s += Qn1[(size_t)n * DM + d];
    qpool[d] = s * (1.f / Nn);
}

// Kernel 3: hyper GEMV -- the HBM roofline floor (streams 256MB fp32 weights
// per matrix). One wave per output element; per-lane float4 -> b128 loads.
// Writes Omega directly as B-frags [DM x DS].
__global__ void hyper_kernel(const float* __restrict__ qpool,
                             const float* __restrict__ W,
                             _Float16* __restrict__ OmB) {
    int wid  = blockIdx.x * 8 + (threadIdx.x >> 5);   // 0 .. DM*DS-1
    int lane = threadIdx.x & 31;
    const float4* wr = (const float4*)(W + (size_t)wid * DM);
    const float4* qp = (const float4*)qpool;
    float s = 0.f;
    for (int it = 0; it < DM / 128; ++it) {
        float4 a = qp[lane + it * 32];
        float4 w4 = wr[lane + it * 32];
        s += a.x * w4.x + a.y * w4.y + a.z * w4.z + a.w * w4.w;
    }
    for (int o = 16; o > 0; o >>= 1) s += __shfl_xor(s, o, 32);
    if (lane == 0) {
        int d = wid >> 6, e = wid & 63;               // row = d*DS + e
        write_bfrag(OmB, d, e, DS / 16, s);
    }
}

// ---------------------------------------------------------------------------
// Generic WMMA GEMM, 2x2 register-blocked: each wave owns a 32x32 output
// block (4 accumulators). Per K-step: 2 A-frags + 2 B-frags (4KB) feed
// 4 WMMAs (64K FLOP) -> 16 FLOP/byte from L2.
// EP epilogues: 0 plain f32 | 1 elu+bias, f32 + A-frag | 2 elu+bias, f32 +
// transposed A-frag | 3 B-frag only | 4 m-proj/Q_interact | 5 add + store out
// ---------------------------------------------------------------------------
template <int EP>
__global__ void gemm_wmma(const _Float16* __restrict__ Af,
                          const _Float16* __restrict__ Bf,
                          int Mtiles, int Ktiles, int Ntiles,
                          float* __restrict__ outF,
                          _Float16* __restrict__ outFrag, int fragT,
                          const float* __restrict__ bias,
                          const float* __restrict__ auxA,
                          const float* __restrict__ auxB,
                          const float* __restrict__ sc0,
                          const float* __restrict__ sc1,
                          int ldo) {
    const int Msup = Mtiles >> 1, Nsup = Ntiles >> 1;
    int wid  = blockIdx.x * (blockDim.x >> 5) + (threadIdx.x >> 5);
    int lane = threadIdx.x & 31;
    if (wid >= Msup * Nsup) return;               // wave-uniform: EXEC stays all-1
    int mt0 = (wid / Nsup) << 1;
    int nt0 = (wid % Nsup) << 1;
    v8f a00 = {}, a01 = {}, a10 = {}, a11 = {};
    for (int kt = 0; kt < Ktiles; ++kt) {
        v16h fa0 = load_frag(Af, (size_t)(mt0 + 0) * Ktiles + kt, lane);
        v16h fa1 = load_frag(Af, (size_t)(mt0 + 1) * Ktiles + kt, lane);
        v16h fb0 = load_frag(Bf, (size_t)kt * Ntiles + nt0 + 0, lane);
        v16h fb1 = load_frag(Bf, (size_t)kt * Ntiles + nt0 + 1, lane);
        a00 = WMMA_F16(fa0, fb0, a00);
        a01 = WMMA_F16(fa0, fb1, a01);
        a10 = WMMA_F16(fa1, fb0, a10);
        a11 = WMMA_F16(fa1, fb1, a11);
    }
    int hi = lane >> 4, cl = lane & 15;
    float dt_s = 0.f, lam_s = 0.f;
    if constexpr (EP == 4) {
        dt_s  = log1pf(__expf(sc0[0]));
        lam_s = log1pf(__expf(sc1[0]));
    }
    auto epi = [&](int mt, int nt, v8f acc) {
        int col = nt * 16 + cl;
        for (int i = 0; i < 8; ++i) {
            int m = mt * 16 + i + hi * 8;
            float x = acc[i];
            if constexpr (EP == 0) {
                outF[(size_t)m * ldo + col] = x;
            } else if constexpr (EP == 1 || EP == 2) {
                float xb = x + bias[col];
                float y  = (xb > 0.f) ? (xb + 1.f) : __expf(xb);   // elu(x)+1
                outF[(size_t)m * ldo + col] = y;
                if constexpr (EP == 1) write_afrag(outFrag, m, col, fragT, y);
                else                   write_afrag(outFrag, col, m, fragT, y);
            } else if constexpr (EP == 3) {
                write_bfrag(outFrag, m, col, fragT, x);
            } else if constexpr (EP == 4) {
                float qi = auxA[(size_t)m * ldo + col];
                float xv = auxB[(size_t)m * ldo + col];
                outF[(size_t)m * ldo + col] = qi + dt_s * (x + bias[col]) + lam_s * (xv - qi);
            } else if constexpr (EP == 5) {
                outF[(size_t)m * ldo + col] = auxA[(size_t)m * ldo + col] + x;
            }
        }
    };
    epi(mt0 + 0, nt0 + 0, a00);
    epi(mt0 + 0, nt0 + 1, a01);
    epi(mt0 + 1, nt0 + 0, a10);
    epi(mt0 + 1, nt0 + 1, a11);
}

// Gated-MLP GEMM: mt-blocked x dual (G/U) accumulators -> 4 WMMAs per K-step
// sharing 2 A-frags and 2 B-frags. H = silu(G)*U scattered straight into
// A-frag layout for the W_down GEMM.
__global__ void gemm_wmma_dual(const _Float16* __restrict__ Af,
                               const _Float16* __restrict__ Bf,
                               _Float16* __restrict__ Hfrag) {
    const int Mtiles = Nn / 16, Ktiles = DM / 32;
    const int NtHalf = INNER / 16, NtFull = G2 / 16, HKt = INNER / 32;
    const int Msup = Mtiles >> 1;
    int wid  = blockIdx.x * 8 + (threadIdx.x >> 5);
    int lane = threadIdx.x & 31;
    if (wid >= Msup * NtHalf) return;
    int mt0 = (wid / NtHalf) << 1;
    int nt  = wid % NtHalf;
    v8f g0 = {}, g1 = {}, u0 = {}, u1 = {};
    for (int kt = 0; kt < Ktiles; ++kt) {
        v16h fa0 = load_frag(Af, (size_t)(mt0 + 0) * Ktiles + kt, lane);
        v16h fa1 = load_frag(Af, (size_t)(mt0 + 1) * Ktiles + kt, lane);
        v16h fbg = load_frag(Bf, (size_t)kt * NtFull + nt, lane);
        v16h fbu = load_frag(Bf, (size_t)kt * NtFull + nt + NtHalf, lane);
        g0 = WMMA_F16(fa0, fbg, g0);
        g1 = WMMA_F16(fa1, fbg, g1);
        u0 = WMMA_F16(fa0, fbu, u0);
        u1 = WMMA_F16(fa1, fbu, u1);
    }
    int hi = lane >> 4, cl = lane & 15;
    int c = nt * 16 + cl;
    for (int mi = 0; mi < 2; ++mi) {
        v8f gg = mi ? g1 : g0;
        v8f uu = mi ? u1 : u0;
        for (int i = 0; i < 8; ++i) {
            int m = (mt0 + mi) * 16 + i + hi * 8;
            float g = gg[i], u = uu[i];
            float h = g * (1.f / (1.f + __expf(-g))) * u;   // silu(g)*u
            write_afrag(Hfrag, m, c, HKt, h);
        }
    }
}

// Kernel: phi_k_sum[e] = sum_n Phi_K[n,e]
__global__ void phiksum_kernel(const float* __restrict__ PhiK, float* __restrict__ pks) {
    int e = threadIdx.x;  // 64 threads
    float s = 0.f;
    for (int n = 0; n < Nn; ++n) s += PhiK[(size_t)n * DS + e];
    pks[e] = s;
}

// Kernel: m = Attraction/Norm - Q_norm1, written as A-frags
__global__ void m_kernel(const float* __restrict__ PhiQ,
                         const float* __restrict__ pks,
                         const float* __restrict__ Attr,
                         const float* __restrict__ Qn1,
                         _Float16* __restrict__ mA) {
    int n = blockIdx.x, t = threadIdx.x;
    float den = 0.f;
    for (int e = 0; e < DS; ++e) den += PhiQ[(size_t)n * DS + e] * pks[e];
    float inv = 1.f / (fabsf(den) + 1.f);
    for (int i = 0; i < 4; ++i) {
        int d = t + i * 256;
        float v = Attr[(size_t)n * DM + d] * inv - Qn1[(size_t)n * DM + d];
        write_afrag(mA, n, d, DM / 32, v);
    }
}

// Kernel: LayerNorm2 -> A-frags
__global__ void ln2_kernel(const float* __restrict__ Qint,
                           const float* __restrict__ gw,
                           const float* __restrict__ gb,
                           _Float16* __restrict__ Qn2A) {
    __shared__ float red[256];
    int n = blockIdx.x, t = threadIdx.x;
    const float* src = Qint + (size_t)n * DM;
    float x[4], s = 0.f;
    for (int i = 0; i < 4; ++i) { x[i] = src[t + i * 256]; s += x[i]; }
    red[t] = s; __syncthreads();
    for (int o = 128; o > 0; o >>= 1) { if (t < o) red[t] += red[t + o]; __syncthreads(); }
    float mu = red[0] * (1.f / DM); __syncthreads();
    float s2 = 0.f;
    for (int i = 0; i < 4; ++i) { float d = x[i] - mu; s2 += d * d; }
    red[t] = s2; __syncthreads();
    for (int o = 128; o > 0; o >>= 1) { if (t < o) red[t] += red[t + o]; __syncthreads(); }
    float inv = rsqrtf(red[0] * (1.f / DM) + 1e-5f);
    for (int i = 0; i < 4; ++i) {
        int d = t + i * 256;
        float y = (x[i] - mu) * inv * gw[d] + gb[d];
        write_afrag(Qn2A, n, d, DM / 32, y);
    }
}

// Kernel: depthwise conv over N (K=3, pad=1), frag-layout in and out
__global__ void dwconv_kernel(const _Float16* __restrict__ Hf,
                              const float* __restrict__ w,
                              _Float16* __restrict__ HcA) {
    int idx = blockIdx.x * 256 + threadIdx.x;   // n*INNER + c
    int c = idx & (INNER - 1), n = idx >> 12;
    const int Kt = INNER / 32;
    float a0 = (n > 0)      ? read_afrag(Hf, n - 1, c, Kt) : 0.f;
    float a1 =                read_afrag(Hf, n,     c, Kt);
    float a2 = (n < Nn - 1) ? read_afrag(Hf, n + 1, c, Kt) : 0.f;
    float o = a0 * w[c * 3 + 0] + a1 * w[c * 3 + 1] + a2 * w[c * 3 + 2];
    write_afrag(HcA, n, c, Kt, o);
}

// Kernel: pack fp32 weight W[o, k] (out-major) into B-frags of W^T [K x N]
__global__ void pack_b_w(const float* __restrict__ W, _Float16* __restrict__ dst,
                         int Ktot, int Ntot) {
    long idx = (long)blockIdx.x * 256 + threadIdx.x;
    if (idx >= (long)Ktot * Ntot) return;
    int k = (int)(idx % Ktot);
    int n = (int)(idx / Ktot);
    write_bfrag(dst, k, n, Ntot >> 4, W[(size_t)n * Ktot + k]);
}

// ---------------------------------------------------------------------------
extern "C" void kernel_launch(void* const* d_in, const int* in_sizes, int n_in,
                              void* d_out, int out_size, void* d_ws, size_t ws_size,
                              hipStream_t stream) {
    const float* Q_in     = (const float*)d_in[0];
    const float* X        = (const float*)d_in[1];
    const float* cosT     = (const float*)d_in[2];
    const float* sinT     = (const float*)d_in[3];
    const float* m_proj_w = (const float*)d_in[4];
    const float* m_proj_b = (const float*)d_in[5];
    const float* hyper_qw = (const float*)d_in[6];
    const float* hyper_kw = (const float*)d_in[7];
    const float* B_Q      = (const float*)d_in[8];
    const float* B_K      = (const float*)d_in[9];
    const float* W_up     = (const float*)d_in[10];
    const float* dw_w     = (const float*)d_in[11];
    const float* W_down   = (const float*)d_in[12];
    const float* n1g      = (const float*)d_in[13];
    const float* n1b      = (const float*)d_in[14];
    const float* n2g      = (const float*)d_in[15];
    const float* n2b      = (const float*)d_in[16];
    const float* dt       = (const float*)d_in[17];
    const float* lam      = (const float*)d_in[18];
    float* out = (float*)d_out;

    // workspace carving (per-batch buffers reused across the 4 batches)
    char* ws = (char*)d_ws; size_t off = 0;
    auto carve = [&](size_t bytes) -> void* {
        void* p = ws + off; off = (off + bytes + 255) & ~(size_t)255; return p;
    };
    float*     Qn1   = (float*)    carve((size_t)Nn * DM * 4);
    _Float16*  Vb    = (_Float16*) carve((size_t)Nn * DM * 2);
    _Float16*  QrotA = (_Float16*) carve((size_t)Nn * DM * 2);
    float*     qpool = (float*)    carve(DM * 4);
    _Float16*  OmQb  = (_Float16*) carve((size_t)DM * DS * 2);
    _Float16*  OmKb  = (_Float16*) carve((size_t)DM * DS * 2);
    float*     PhiQ  = (float*)    carve((size_t)Nn * DS * 4);
    float*     PhiK  = (float*)    carve((size_t)Nn * DS * 4);
    _Float16*  PhiQA = (_Float16*) carve((size_t)Nn * DS * 2);
    _Float16*  PhiKT = (_Float16*) carve((size_t)Nn * DS * 2);
    float*     pks   = (float*)    carve(DS * 4);
    _Float16*  Cb    = (_Float16*) carve((size_t)DS * DM * 2);
    float*     Attr  = (float*)    carve((size_t)Nn * DM * 4);
    _Float16*  mA    = (_Float16*) carve((size_t)Nn * DM * 2);
    float*     Qint  = (float*)    carve((size_t)Nn * DM * 4);
    _Float16*  Qn2A  = (_Float16*) carve((size_t)Nn * DM * 2);
    _Float16*  Hfrag = (_Float16*) carve((size_t)Nn * INNER * 2);
    _Float16*  HcA   = (_Float16*) carve((size_t)Nn * INNER * 2);
    _Float16*  mprjB = (_Float16*) carve((size_t)DM * DM * 2);
    _Float16*  WupB  = (_Float16*) carve((size_t)DM * G2 * 2);
    _Float16*  WdnB  = (_Float16*) carve((size_t)INNER * DM * 2);
    (void)ws_size; (void)n_in; (void)in_sizes; (void)out_size;

    // batch-independent weight packing (f32 -> f16 B-fragments)
    pack_b_w<<<(DM * DM) / 256, 256, 0, stream>>>(m_proj_w, mprjB, DM, DM);
    pack_b_w<<<(DM * G2) / 256, 256, 0, stream>>>(W_up, WupB, DM, G2);
    pack_b_w<<<(INNER * DM) / 256, 256, 0, stream>>>(W_down, WdnB, INNER, DM);

    for (int b = 0; b < Bn; ++b) {
        const size_t base = (size_t)b * Nn * DM;
        ln_rope_kernel<<<Nn, 256, 0, stream>>>(Q_in + base, cosT, sinT, n1g, n1b,
                                               Qn1, Vb, QrotA);
        pool_kernel<<<DM / 256, 256, 0, stream>>>(Qn1, qpool);
        hyper_kernel<<<(DM * DS) / 8, 256, 0, stream>>>(qpool, hyper_qw, OmQb);
        hyper_kernel<<<(DM * DS) / 8, 256, 0, stream>>>(qpool, hyper_kw, OmKb);
        // Phi_Q = elu(Qrot @ Omega_Q + B_Q)+1   [4096x64]: 128x2 super-tiles
        gemm_wmma<1><<<(128 * 2) / 8, 256, 0, stream>>>(QrotA, OmQb, 256, 32, 4,
            PhiQ, PhiQA, DS / 32, B_Q, nullptr, nullptr, nullptr, nullptr, DS);
        // Phi_K -> f32 + transposed A-frag (Phi_K^T is the A of the C GEMM)
        gemm_wmma<2><<<(128 * 2) / 8, 256, 0, stream>>>(QrotA, OmKb, 256, 32, 4,
            PhiK, PhiKT, Nn / 32, B_K, nullptr, nullptr, nullptr, nullptr, DS);
        phiksum_kernel<<<1, 64, 0, stream>>>(PhiK, pks);
        // C = Phi_K^T @ V   [64x1024], stored directly as B-frags
        gemm_wmma<3><<<(2 * 32) / 8, 256, 0, stream>>>(PhiKT, Vb, 4, 128, 64,
            nullptr, Cb, DM / 16, nullptr, nullptr, nullptr, nullptr, nullptr, 0);
        // Attraction = Phi_Q @ C   [4096x1024]
        gemm_wmma<0><<<(128 * 32) / 8, 256, 0, stream>>>(PhiQA, Cb, 256, 2, 64,
            Attr, nullptr, 0, nullptr, nullptr, nullptr, nullptr, nullptr, DM);
        m_kernel<<<Nn, 256, 0, stream>>>(PhiQ, pks, Attr, Qn1, mA);
        // Q_interact = Q_in + softplus(dt)*(m@Wp^T+b) + softplus(lam)*(X-Q_in)
        gemm_wmma<4><<<(128 * 32) / 8, 256, 0, stream>>>(mA, mprjB, 256, 32, 64,
            Qint, nullptr, 0, m_proj_b, Q_in + base, X + base, dt, lam, DM);
        ln2_kernel<<<Nn, 256, 0, stream>>>(Qint, n2g, n2b, Qn2A);
        // H = silu(G)*U, mt-blocked dual GEMM against W_up^T [1024x8192]
        gemm_wmma_dual<<<(128 * 256) / 8, 256, 0, stream>>>(Qn2A, WupB, Hfrag);
        dwconv_kernel<<<(Nn * INNER) / 256, 256, 0, stream>>>(Hfrag, dw_w, HcA);
        // out = Q_interact + H_conv @ W_down^T
        gemm_wmma<5><<<(128 * 32) / 8, 256, 0, stream>>>(HcA, WdnB, 256, 128, 64,
            out + base, nullptr, 0, nullptr, Qint, nullptr, nullptr, nullptr, DM);
    }
}
